// Tno_64510408786206
// MI455X (gfx1250) — compile-verified, matrix operand order checked by reading, not run on hardware
//
#include <hip/hip_runtime.h>

#define Bb 4
#define Nn 4096
#define Dd 1024
#define Ff 64
#define KTAPS 256

typedef __attribute__((ext_vector_type(16))) __bf16 v16bf;
typedef __attribute__((ext_vector_type(8)))  float  v8f;
typedef __attribute__((ext_vector_type(4)))  unsigned int v4u;
typedef __attribute__((ext_vector_type(8)))  int    v8i;
typedef __attribute__((ext_vector_type(4)))  int    v4i;

#if defined(__gfx1250__) && __has_builtin(__builtin_amdgcn_tensor_load_to_lds)
#define USE_TDM 1
#else
#define USE_TDM 0
#endif

__device__ __forceinline__ float silu_f(float t) { return t / (1.0f + __expf(-t)); }

// -------------------------------------------------------------------------
// Kernel A: RPE MLP for rows k = 0..KTAPS-1 (rows beyond KTAPS are killed by
// the exp-decay). 128 threads = 4 wave32; each wave owns a 16-row strip and
// does the 64x64 GEMM with v_wmma_f32_16x16x32_bf16.
// -------------------------------------------------------------------------
__global__ void rpe_mlp_kernel(const float* __restrict__ rpe_in_w,
                               const float* __restrict__ lns_g,
                               const float* __restrict__ lns_b,
                               const float* __restrict__ mid_w,
                               const float* __restrict__ out_ln_g,
                               const float* __restrict__ out_ln_b,
                               float* __restrict__ u) {
  __shared__ float hs[64][Ff + 1];
  __shared__ float av[64][Ff + 1];
  const int tid  = threadIdx.x;      // 128
  const int lane = tid & 31;
  const int wave = tid >> 5;
  const int hi   = lane >> 4;        // which half-wave (K split)
  const int ml   = lane & 15;        // row (A) / col (B,C) within tile
  const int row0 = blockIdx.x * 64;

  // h[r][f] = index * rpe_in_w[f]
  for (int i = tid; i < 64 * Ff; i += 128) {
    int r = i >> 6, f = i & 63;
    hs[r][f] = (float)(row0 + r) * rpe_in_w[f];
  }
  __syncthreads();

  for (int layer = 0; layer < 3; ++layer) {
    // LN + silu (one row per thread, f32)
    if (tid < 64) {
      float mu = 0.f;
      for (int f = 0; f < Ff; ++f) mu += hs[tid][f];
      mu *= (1.0f / Ff);
      float var = 0.f;
      for (int f = 0; f < Ff; ++f) { float d = hs[tid][f] - mu; var += d * d; }
      var *= (1.0f / Ff);
      const float rs = rsqrtf(var + 1e-5f);
      for (int f = 0; f < Ff; ++f) {
        float t = (hs[tid][f] - mu) * rs * lns_g[layer * Ff + f] + lns_b[layer * Ff + f];
        av[tid][f] = silu_f(t);
      }
    }
    __syncthreads();

    // h += a @ mid_w[layer].T  (M=16/wave, N=64 in 4 tiles, K=64 in 2 steps)
    const int m0 = wave * 16;
    for (int nt = 0; nt < 4; ++nt) {
      v8f acc = {};
      for (int kt = 0; kt < 2; ++kt) {
        const int koff = kt * 32 + hi * 8;
        v16bf af, bf;
        const float* wrow = &mid_w[(size_t)layer * Ff * Ff + (size_t)(nt * 16 + ml) * Ff];
#pragma unroll
        for (int j = 0; j < 8; ++j) {
          const int kb = koff + ((j < 4) ? (2 * j) : (16 + 2 * (j - 4)));
          af[2 * j]     = (__bf16)av[m0 + ml][kb];
          af[2 * j + 1] = (__bf16)av[m0 + ml][kb + 1];
          bf[2 * j]     = (__bf16)wrow[kb];       // B[k,n] = mid_w[n,k]
          bf[2 * j + 1] = (__bf16)wrow[kb + 1];
        }
        acc = __builtin_amdgcn_wmma_f32_16x16x32_bf16(false, af, false, bf,
                                                      (short)0, acc, false, false);
      }
#pragma unroll
      for (int r = 0; r < 8; ++r)
        hs[m0 + r + 8 * hi][nt * 16 + ml] += acc[r];   // residual add, own rows only
    }
    __syncthreads();
  }

  // final LN + silu -> u[k][f]
  if (tid < 64) {
    float mu = 0.f;
    for (int f = 0; f < Ff; ++f) mu += hs[tid][f];
    mu *= (1.0f / Ff);
    float var = 0.f;
    for (int f = 0; f < Ff; ++f) { float d = hs[tid][f] - mu; var += d * d; }
    var *= (1.0f / Ff);
    const float rs = rsqrtf(var + 1e-5f);
    const int row = row0 + tid;
    for (int f = 0; f < Ff; ++f) {
      float t = (hs[tid][f] - mu) * rs * out_ln_g[f] + out_ln_b[f];
      u[(size_t)row * Ff + f] = silu_f(t);
    }
  }
}

// -------------------------------------------------------------------------
// Kernel B: wdec[k][d] = exp(k * logsigmoid(gamma[d])) * (u[k] . out_w[d])
// GEMM (KTAPS x 64) @ (64 x 1024) via bf16 WMMA; decay fused; k-major store
// so the FIR kernel's tap tile is a contiguous-row 2D tile for the TDM.
// -------------------------------------------------------------------------
__global__ void wproj_kernel(const float* __restrict__ u,
                             const float* __restrict__ out_w,
                             const float* __restrict__ gamma,
                             float* __restrict__ wdec) {
  const int tid  = threadIdx.x;      // 128 = 4 waves, one 16x16 tile each
  const int lane = tid & 31;
  const int wave = tid >> 5;
  const int hi   = lane >> 4;
  const int ml   = lane & 15;
  const int idx  = blockIdx.x * 4 + wave;     // 0..(KTAPS/16)*(Dd/16)-1
  const int k0   = (idx >> 6) * 16;
  const int d0   = (idx & 63) * 16;

  v8f acc = {};
  for (int kt = 0; kt < 2; ++kt) {
    const int koff = kt * 32 + hi * 8;
    v16bf af, bf;
    const float* arow = &u[(size_t)(k0 + ml) * Ff];
    const float* brow = &out_w[(size_t)(d0 + ml) * Ff];   // B[f,d] = out_w[d,f]
#pragma unroll
    for (int j = 0; j < 8; ++j) {
      const int kb = koff + ((j < 4) ? (2 * j) : (16 + 2 * (j - 4)));
      af[2 * j]     = (__bf16)arow[kb];
      af[2 * j + 1] = (__bf16)arow[kb + 1];
      bf[2 * j]     = (__bf16)brow[kb];
      bf[2 * j + 1] = (__bf16)brow[kb + 1];
    }
    acc = __builtin_amdgcn_wmma_f32_16x16x32_bf16(false, af, false, bf,
                                                  (short)0, acc, false, false);
  }
  const int d = d0 + ml;
  const float g = gamma[d];
  const float lam = -log1pf(__expf(-g));      // logsigmoid(gamma)
#pragma unroll
  for (int r = 0; r < 8; ++r) {
    const int k = k0 + r + 8 * hi;
    wdec[(size_t)k * Dd + d] = acc[r] * __expf((float)k * lam);
  }
}

// -------------------------------------------------------------------------
// Kernel C: 256-tap causal FIR, register sliding window.
// y[b,t,d] = sum_{k<=min(t,255)} wdec[k][d] * x[b,t-k,d]
// The per-block 64ch x 256tap weight tile (64 KB) is staged into LDS once
// via the Tensor Data Mover (tensor_load_to_lds + s_wait_tensorcnt), then
// the FMA loop reads taps from LDS (conflict-free: 32 lanes -> 32 banks).
// Thread owns 8 consecutive t for one channel: per tap 1 ds_load + 1 global
// x load + 8 FMAs. x (64 MB) stays resident in the 192 MB L2.
// -------------------------------------------------------------------------
__global__ void fir_kernel(const float* __restrict__ x,
                           const float* __restrict__ wdec,
                           float* __restrict__ y) {
  __shared__ float wlds[KTAPS * 64];
  const int tid = threadIdx.x;              // 256
  const unsigned bid = blockIdx.x;          // 8192 = 4 * 128 * 16
  const int dt = bid & 15;
  const int tt = (bid >> 4) & 127;
  const int b  = bid >> 11;
  const int d0 = dt * 64;
  const int dl = tid & 63;
  const int d  = d0 + dl;
  const int tbase = tt * 32 + (tid >> 6) * 8;
  const float* xb = x + (size_t)b * Nn * Dd + d;

#if USE_TDM
  if (tid < 32) {                           // one TDM issue per block (wave 0)
    const unsigned long long ga = (unsigned long long)(uintptr_t)(wdec + d0);
    const unsigned int lab = (unsigned int)(uintptr_t)(&wlds[0]);
    v4u g0;
    g0.x = 1u;                                              // count=1 user D#
    g0.y = lab;                                             // lds_addr
    g0.z = (unsigned int)ga;                                // global_addr[31:0]
    g0.w = (unsigned int)((ga >> 32) & 0x01FFFFFFu) | 0x80000000u; // [56:32]|type=2
    v8i g1;
    g1[0] = (int)(2u << 16);        // data_size = 4 bytes, no multicast/pad
    g1[1] = (int)((unsigned)Dd << 16);   // tensor_dim0 = 1024 (bits 79:48)
    g1[2] = (int)((unsigned)KTAPS << 16);// dim0_hi=0 | tensor_dim1 = 256
    g1[3] = (int)(64u << 16);       // dim1_hi=0 | tile_dim0 = 64
    g1[4] = (int)KTAPS;             // tile_dim1 = 256, tile_dim2 = 0
    g1[5] = Dd;                     // tensor_dim0_stride = 1024 (low 32)
    g1[6] = 0;
    g1[7] = 0;
    v4i gz = {0, 0, 0, 0};
#if __clang_major__ >= 23
    v8i gz8 = {0, 0, 0, 0, 0, 0, 0, 0};
    __builtin_amdgcn_tensor_load_to_lds(g0, g1, gz, gz, gz8, 0);
#else
    __builtin_amdgcn_tensor_load_to_lds(g0, g1, gz, gz, 0);
#endif
    __builtin_amdgcn_s_wait_tensorcnt(0);
  }
  __syncthreads();
#else
  for (int i = tid; i < KTAPS * 64; i += 256)
    wlds[i] = wdec[(size_t)(i >> 6) * Dd + d0 + (i & 63)];
  __syncthreads();
#endif

  float acc[8] = {0.f, 0.f, 0.f, 0.f, 0.f, 0.f, 0.f, 0.f};
  float xw[8];
#pragma unroll
  for (int i = 0; i < 8; ++i) xw[i] = xb[(size_t)(tbase + i) * Dd];

#pragma unroll 8
  for (int k = 0; k < KTAPS; ++k) {
    const float wv = wlds[k * 64 + dl];
    if ((k & 31) == 0) {
      const int tp = tbase - k - 32;
      if (tp >= 0) __builtin_prefetch(&xb[(size_t)tp * Dd], 0, 1);
    }
#pragma unroll
    for (int i = 0; i < 8; ++i) acc[i] = fmaf(wv, xw[i], acc[i]);
#pragma unroll
    for (int i = 7; i > 0; --i) xw[i] = xw[i - 1];    // slide window down in t
    const int ts = tbase - k - 1;
    xw[0] = (ts >= 0) ? xb[(size_t)ts * Dd] : 0.0f;   // zero-fill = causality
  }

  float* yb = y + (size_t)b * Nn * Dd + d;
#pragma unroll
  for (int i = 0; i < 8; ++i) yb[(size_t)(tbase + i) * Dd] = acc[i];
}

// -------------------------------------------------------------------------
// Kernel D: in-place channel LayerNorm over D=1024 per (b,t) row.
// -------------------------------------------------------------------------
__global__ void ln_kernel(float* __restrict__ y,
                          const float* __restrict__ g,
                          const float* __restrict__ bta) {
  __shared__ float s1[256];
  __shared__ float s2[256];
  const int tid = threadIdx.x;
  float* p = y + (size_t)blockIdx.x * Dd;
  float v[4];
  float sum = 0.f, sq = 0.f;
#pragma unroll
  for (int j = 0; j < 4; ++j) {
    v[j] = p[tid + j * 256];
    sum += v[j];
    sq += v[j] * v[j];
  }
  s1[tid] = sum; s2[tid] = sq;
  __syncthreads();
  for (int off = 128; off > 0; off >>= 1) {
    if (tid < off) { s1[tid] += s1[tid + off]; s2[tid] += s2[tid + off]; }
    __syncthreads();
  }
  const float mu  = s1[0] * (1.0f / Dd);
  const float var = fmaxf(s2[0] * (1.0f / Dd) - mu * mu, 0.f);
  const float rs  = rsqrtf(var + 1e-5f);
#pragma unroll
  for (int j = 0; j < 4; ++j) {
    const int c = tid + j * 256;
    p[c] = (v[j] - mu) * rs * g[c] + bta[c];
  }
}

extern "C" void kernel_launch(void* const* d_in, const int* in_sizes, int n_in,
                              void* d_out, int out_size, void* d_ws, size_t ws_size,
                              hipStream_t stream) {
  const float* x        = (const float*)d_in[0];
  const float* rpe_in_w = (const float*)d_in[1];
  const float* lns_g    = (const float*)d_in[2];
  const float* lns_b    = (const float*)d_in[3];
  const float* mid_w    = (const float*)d_in[4];
  const float* out_ln_g = (const float*)d_in[5];
  const float* out_ln_b = (const float*)d_in[6];
  const float* out_w    = (const float*)d_in[7];
  const float* gamma    = (const float*)d_in[8];
  const float* norm_g   = (const float*)d_in[9];
  const float* norm_b   = (const float*)d_in[10];
  float* out = (float*)d_out;

  float* u    = (float*)d_ws;                 // KTAPS * 64 f32
  float* wdec = u + (size_t)KTAPS * Ff;       // KTAPS * Dd f32  (~1 MB total)

  rpe_mlp_kernel<<<KTAPS / 64, 128, 0, stream>>>(rpe_in_w, lns_g, lns_b, mid_w,
                                                 out_ln_g, out_ln_b, u);
  wproj_kernel<<<(KTAPS / 16) * (Dd / 16) / 4, 128, 0, stream>>>(u, out_w, gamma, wdec);
  fir_kernel<<<Bb * (Nn / 32) * (Dd / 64), 256, 0, stream>>>(x, wdec, out);
  ln_kernel<<<Bb * Nn, 256, 0, stream>>>(out, norm_g, norm_b);
}
